// MinkBEVBackbone_57750130262651
// MI455X (gfx1250) — matrix operand order repaired and checked by reading, not compiled, not run on hardware
//
#include <hip/hip_runtime.h>

// CDNA5 (gfx1250) wave32 WMMA types
typedef __attribute__((ext_vector_type(16))) __bf16 v16bf;
typedef __attribute__((ext_vector_type(8)))  float  v8f;
// TDM descriptor groups (amdgpu-toolchain / clang-23 6-arg builtin form)
typedef __attribute__((ext_vector_type(4))) unsigned int u32x4;
typedef __attribute__((ext_vector_type(8))) int          i32x8;
typedef __attribute__((ext_vector_type(4))) int          i32x4;

#define BN_EPS 1e-5f
#define NT 4   // N-tiles (16 cols each) per wave -> 16x64 output strip per wave

// ---------------------------------------------------------------------------
// Elementwise: y = x * mask (mask broadcast over channels). Pow2 shift decode.
// ---------------------------------------------------------------------------
__global__ void k_mask_mul(const float* __restrict__ x, const float* __restrict__ m,
                           float* __restrict__ y, int lgC, int lgHW, int total) {
  int i = blockIdx.x * blockDim.x + threadIdx.x;
  if (i >= total) return;
  int p = i & ((1 << lgHW) - 1);
  int b = i >> (lgC + lgHW);
  y[i] = x[i] * m[(b << lgHW) + p];
}

// ---------------------------------------------------------------------------
// Sum of a buffer (cnt = sum(mask)); LDS tree + global f32 atomic
// ---------------------------------------------------------------------------
__global__ void k_sum(const float* __restrict__ m, float* __restrict__ out, int n) {
  __shared__ float sh[256];
  float s = 0.f;
  for (int i = blockIdx.x * blockDim.x + threadIdx.x; i < n; i += gridDim.x * blockDim.x)
    s += m[i];
  sh[threadIdx.x] = s;
  __syncthreads();
  for (int o = 128; o > 0; o >>= 1) {
    if (threadIdx.x < o) sh[threadIdx.x] += sh[threadIdx.x + o];
    __syncthreads();
  }
  if (threadIdx.x == 0) atomicAdd(out, sh[0]);
}

// ---------------------------------------------------------------------------
// Masked BN statistics: stats[2c] = sum(x*m), stats[2c+1] = sum(x*x*m)
// ---------------------------------------------------------------------------
__global__ void k_bn_stats(const float* __restrict__ x, const float* __restrict__ m,
                           float* __restrict__ stats, int B, int lgC, int lgHW) {
  int c = blockIdx.y;
  __shared__ float s1[256], s2[256];
  float a1 = 0.f, a2 = 0.f;
  int n = B << lgHW;
  for (int i = blockIdx.x * blockDim.x + threadIdx.x; i < n; i += gridDim.x * blockDim.x) {
    int b = i >> lgHW, p = i & ((1 << lgHW) - 1);
    float mv = m[i];
    float v  = x[(((size_t)((b << lgC) + c)) << lgHW) + p];
    a1 += v * mv;
    a2 += v * v * mv;
  }
  s1[threadIdx.x] = a1; s2[threadIdx.x] = a2;
  __syncthreads();
  for (int o = 128; o > 0; o >>= 1) {
    if (threadIdx.x < o) { s1[threadIdx.x] += s1[threadIdx.x + o]; s2[threadIdx.x] += s2[threadIdx.x + o]; }
    __syncthreads();
  }
  if (threadIdx.x == 0) {
    atomicAdd(&stats[2 * c],     s1[0]);
    atomicAdd(&stats[2 * c + 1], s2[0]);
  }
}

// ---------------------------------------------------------------------------
// Fused BN normalize + gamma/beta + mask (+ residual) (+ ReLU)
// ---------------------------------------------------------------------------
__global__ void k_bn_apply(const float* __restrict__ x, const float* __restrict__ m,
                           const float* __restrict__ stats, const float* __restrict__ cnt,
                           const float* __restrict__ gamma, const float* __restrict__ beta,
                           const float* __restrict__ resid, float* __restrict__ y,
                           int lgC, int lgHW, int relu, int total) {
  int i = blockIdx.x * blockDim.x + threadIdx.x;
  if (i >= total) return;
  int p = i & ((1 << lgHW) - 1);
  int c = (i >> lgHW) & ((1 << lgC) - 1);
  int b = i >> (lgHW + lgC);
  float cv   = fmaxf(cnt[0], 1.f);
  float mean = stats[2 * c] / cv;
  float var  = stats[2 * c + 1] / cv - mean * mean;
  float inv  = rsqrtf(var + BN_EPS);
  float mv   = m[(b << lgHW) + p];
  float v = ((x[i] - mean) * inv * gamma[c] + beta[c]) * mv;
  if (resid) v += resid[i];
  if (relu)  v = fmaxf(v, 0.f);
  y[i] = v;
}

// ---------------------------------------------------------------------------
// Masked max-pool k=3 s=2 pad=1: pooled occupancy mask
// ---------------------------------------------------------------------------
__global__ void k_pool_mask(const float* __restrict__ m, float* __restrict__ nm,
                            int H, int W, int lgH, int lgW, int lgHo, int lgWo,
                            int total) {
  int i = blockIdx.x * blockDim.x + threadIdx.x;
  if (i >= total) return;
  int wo = i & ((1 << lgWo) - 1);
  int ho = (i >> lgWo) & ((1 << lgHo) - 1);
  int b  = i >> (lgWo + lgHo);
  float best = 0.f;
  for (int kh = 0; kh < 3; ++kh)
    for (int kw = 0; kw < 3; ++kw) {
      int ih = 2 * ho - 1 + kh, iw = 2 * wo - 1 + kw;
      if ((unsigned)ih < (unsigned)H && (unsigned)iw < (unsigned)W)
        best = fmaxf(best, m[(((b << lgH) + ih) << lgW) + iw]);
    }
  nm[i] = best;
}

// ---------------------------------------------------------------------------
// Masked max-pool of the features
// ---------------------------------------------------------------------------
__global__ void k_pool(const float* __restrict__ x, const float* __restrict__ m,
                       const float* __restrict__ nm, float* __restrict__ y,
                       int H, int W, int lgC, int lgH, int lgW, int lgHo, int lgWo,
                       int total) {
  int i = blockIdx.x * blockDim.x + threadIdx.x;
  if (i >= total) return;
  int wo = i & ((1 << lgWo) - 1);
  int ho = (i >> lgWo) & ((1 << lgHo) - 1);
  int c  = (i >> (lgWo + lgHo)) & ((1 << lgC) - 1);
  int b  = i >> (lgWo + lgHo + lgC);
  float best = -3.402823466e38f;
  for (int kh = 0; kh < 3; ++kh)
    for (int kw = 0; kw < 3; ++kw) {
      int ih = 2 * ho - 1 + kh, iw = 2 * wo - 1 + kw;
      if ((unsigned)ih < (unsigned)H && (unsigned)iw < (unsigned)W &&
          m[(((b << lgH) + ih) << lgW) + iw] > 0.f)
        best = fmaxf(best, x[(((size_t)((b << lgC) + c)) << (lgH + lgW)) +
                             (ih << lgW) + iw]);
    }
  y[i] = (nm[(((b << lgHo) + ho) << lgWo) + wo] > 0.f) ? best : 0.f;
}

// ---------------------------------------------------------------------------
// Final bias + mask
// ---------------------------------------------------------------------------
__global__ void k_bias_mask(const float* __restrict__ x, const float* __restrict__ bias,
                            const float* __restrict__ m, float* __restrict__ y,
                            int lgC, int lgHW, int total) {
  int i = blockIdx.x * blockDim.x + threadIdx.x;
  if (i >= total) return;
  int p = i & ((1 << lgHW) - 1);
  int c = (i >> lgHW) & ((1 << lgC) - 1);
  int b = i >> (lgHW + lgC);
  y[i] = (x[i] + bias[c]) * m[(b << lgHW) + p];
}

// ---------------------------------------------------------------------------
// Implicit-GEMM conv via V_WMMA_F32_16X16X32_BF16 (wave32).
//   GEMM: M = Cout, N = B*H*W, K ordered as (kh, kw, ci) in chunks of 32 ci.
// Block = 8 waves sharing one M-tile (blockIdx.y). The block's entire weight
// slice  W[tmbase..tmbase+16, :]  is one CONTIGUOUS run of 16*Kt f32
// (<=147 KB, fits 320 KB LDS) -> DMA'd once by the Tensor Data Mover
// (tensor_load_to_lds, 1-D D#: data_size=4B, tile_dim0=16*Kt) by wave 0,
// s_wait_tensorcnt 0, one barrier. The K-loop then reads A from LDS and
// gathers B from global with pure stride-HW addressing (no divisions: all
// H/W/C are powers of two -> shift decode). Each wave keeps NT=4 f32
// accumulators (16x64 strip): 4 WMMAs per A tile. No early returns: EXEC is
// all-ones at every WMMA; tail waves compute on zero B and skip the store.
// ---------------------------------------------------------------------------
__global__ void __launch_bounds__(256)
k_conv_wmma(const float* __restrict__ in, const float* __restrict__ wgt,
            float* __restrict__ out,
            int B, int Cin, int Cout, int H, int W, int Kk,
            int lgHW, int lgW) {
  extern __shared__ float lds_w[];   // [16][Kt] weight slice for this M-tile

  const int tid  = threadIdx.x;
  const int lane = tid & 31;
  const int wv   = tid >> 5;
  const int m16  = lane & 15;
  const int half = lane >> 4;
  const int HWl  = 1 << lgHW;
  const int Nsp  = B << lgHW;
  const int KK   = Kk * Kk;
  const int Kt   = Cin * KK;
  const int pad  = Kk >> 1;
  const int tmbase = blockIdx.y << 4;
  const int tnbase = (blockIdx.x * 8 + wv) * NT;

  // ---- stage this M-tile's weights into LDS via the Tensor Data Mover ----
#if __has_builtin(__builtin_amdgcn_tensor_load_to_lds) && \
    __has_builtin(__builtin_amdgcn_s_wait_tensorcnt)
  if (wv == 0) {
    const unsigned nelem = (unsigned)(16 * Kt);   // contiguous f32 elements
    unsigned long long ga =
        (unsigned long long)(const void*)(wgt + (size_t)tmbase * Kt);
    unsigned lds_off = (unsigned)(unsigned long long)(void*)lds_w;
    u32x4 g0;
    g0.x = 1u;                                         // count=1 (valid D#)
    g0.y = lds_off;                                    // lds_addr (bytes)
    g0.z = (unsigned)(ga & 0xffffffffu);               // global_addr[31:0]
    g0.w = (unsigned)((ga >> 32) & 0x01ffffffu)        // global_addr[56:32]
         | (2u << 30);                                 // type=2 ("image")
    i32x8 g1;
    g1[0] = (int)(2u << 16);                           // data_size=2 -> 4 bytes
    g1[1] = (int)((nelem & 0xffffu) << 16);            // tensor_dim0[15:0]
    g1[2] = (int)((nelem >> 16) & 0xffffu);            // tensor_dim0[31:16]
    g1[3] = (int)((nelem & 0xffffu) << 16);            // tile_dim0 (1-D tile)
    g1[4] = 0;                                         // tile_dim1/2 unused
    g1[5] = (int)nelem;                                // tensor_dim0_stride lo
    g1[6] = 0;
    g1[7] = 0;
    i32x4 gz4 = {0, 0, 0, 0};                          // groups 2/3: <=2-D
    i32x8 gz8 = {0, 0, 0, 0, 0, 0, 0, 0};              // extra group (unused)
    __builtin_amdgcn_tensor_load_to_lds(g0, g1, gz4, gz4, gz8, 0);
    __builtin_amdgcn_s_wait_tensorcnt(0);
  }
#else
  for (int e = tid; e < 16 * Kt; e += 256)
    lds_w[e] = wgt[(size_t)tmbase * Kt + e];
#endif
  __syncthreads();

  // Decode this lane's NT output columns once (pure shifts).
  int nb[NT], nh[NT], nw[NT];
  bool nvalid[NT];
#pragma unroll
  for (int t = 0; t < NT; ++t) {
    int ncol = (tnbase + t) * 16 + m16;
    nvalid[t] = ncol < Nsp;
    nb[t] = ncol >> lgHW;
    int r = ncol & (HWl - 1);
    nh[t] = r >> lgW;
    nw[t] = r & ((1 << lgW) - 1);
  }

  const float* wrow = lds_w + (size_t)m16 * Kt;   // this lane's A row (LDS)
  v8f acc[NT] = {};

  for (int kpos = 0; kpos < KK; ++kpos) {
    const int kh = kpos / Kk;                 // outer-loop only (uniform)
    const int kw = kpos - kh * Kk;

    // Per-tile spatial base for this filter tap (SAME padding).
    size_t base[NT];
    bool ok[NT];
#pragma unroll
    for (int t = 0; t < NT; ++t) {
      int ih = nh[t] + kh - pad, iw = nw[t] + kw - pad;
      ok[t] = nvalid[t] && (unsigned)ih < (unsigned)H && (unsigned)iw < (unsigned)W;
      base[t] = (((size_t)nb[t] * Cin) << lgHW) + (size_t)((ih << lgW) + iw);
    }

    for (int cb = 0; cb < Cin; cb += 32) {
      // ---- A from LDS weight slice, 16x32 ISA layout:
      //   element e -> local K  kl = (v<4 ? 2v : 16+2(v-4)) + half*8 + (e&1)
      v16bf a;
#pragma unroll
      for (int e = 0; e < 16; ++e) {
        int v  = e >> 1;
        int kl = ((v < 4) ? (v << 1) : (16 + ((v - 4) << 1))) + (half << 3) + (e & 1);
        int ci = cb + kl;
        float f = (ci < Cin) ? wrow[(size_t)ci * KK + kpos] : 0.f;
        a[e] = (__bf16)f;
      }

      // ---- 4 WMMAs per A tile (B gather: pure stride-HW, no division)
#pragma unroll
      for (int t = 0; t < NT; ++t) {
        const float* bp = in + base[t] + ((size_t)(cb + (half << 3)) << lgHW);
        v16bf bm;
#pragma unroll
        for (int e2 = 0; e2 < 16; ++e2) {
          int ci  = cb + (half << 3) + e2 + (e2 & 8);
          float f = (ok[t] && ci < Cin) ? bp[(size_t)(e2 + (e2 & 8)) << lgHW] : 0.f;
          bm[e2] = (__bf16)f;
        }
        acc[t] = __builtin_amdgcn_wmma_f32_16x16x32_bf16(false, a, false, bm,
                                                         (short)0, acc[t],
                                                         false, false);
      }
    }
  }

  // ---- epilogue: C/D layout (lane n = L%16, VGPR r -> row r + 8*half)
#pragma unroll
  for (int t = 0; t < NT; ++t) {
    if (!nvalid[t]) continue;     // per-lane divergence after all WMMAs: fine
#pragma unroll
    for (int r = 0; r < 8; ++r) {
      int mrow = tmbase + r + half * 8;
      out[(((size_t)(nb[t] * Cout + mrow)) << lgHW) + (nh[t] << lgW) + nw[t]] =
          acc[t][r];
    }
  }
}

// ---------------------------------------------------------------------------
// Host orchestration
// ---------------------------------------------------------------------------
static inline int ilog2i(int v) { int r = 0; while ((1 << r) < v) ++r; return r; }

extern "C" void kernel_launch(void* const* d_in, const int* in_sizes, int n_in,
                              void* d_out, int out_size, void* d_ws, size_t ws_size,
                              hipStream_t stream) {
  (void)in_sizes; (void)n_in; (void)out_size; (void)ws_size;

  // Allow the big weight slices (<=16*2304*4 = 147456 B) as dynamic LDS.
  (void)hipFuncSetAttribute((const void*)k_conv_wmma,
                            hipFuncAttributeMaxDynamicSharedMemorySize, 160 * 1024);

  const float* x     = (const float*)d_in[0];
  const float* mask0 = (const float*)d_in[1];

  // ---- params: jax pytree (dict keys sorted alphabetically) ----
  struct BP {
    const float *b1, *b2, *b3, *bs, *g1, *g2, *g3, *gs, *w1, *w2, *w3, *wsd;
    int cin, mid, cout, k; bool has_s;
  };
  int idx = 2;
  auto nxt = [&]() -> const float* { return (const float*)d_in[idx++]; };
  auto mk = [&](bool s, int cin, int cout, int k) -> BP {
    BP p{}; p.cin = cin; p.cout = cout; p.mid = cout / 4; p.k = k; p.has_s = s;
    p.b1 = nxt(); p.b2 = nxt(); p.b3 = nxt();
    p.bs = s ? nxt() : nullptr;
    p.g1 = nxt(); p.g2 = nxt(); p.g3 = nxt();
    p.gs = s ? nxt() : nullptr;
    p.w1 = nxt(); p.w2 = nxt(); p.w3 = nxt();
    p.wsd = s ? nxt() : nullptr;
    return p;
  };
  BP b1a = mk(true,  32,  64, 11);
  BP b1b = mk(false, 64,  64, 11);
  BP b2a = mk(true,  64, 128, 7);
  BP b2b = mk(false,128, 128, 7);
  BP b3a = mk(true, 128, 256, 5);
  BP b3b = mk(false,256, 256, 5);
  const float* b4b = nxt();
  const float* b4w = nxt();

  // ---- deterministic bump allocator over d_ws ----
  size_t off = 0;
  auto alloc = [&](size_t elems) -> float* {
    float* p = (float*)((char*)d_ws + off);
    off += ((elems * sizeof(float) + 255) & ~(size_t)255);
    return p;
  };
  const int B = 4;
  const int HW0 = 256 * 256, N0 = B * HW0;
  const int HW1 = 128 * 128, N1 = B * HW1;
  const int HW2 =  64 *  64, N2 = B * HW2;
  const int HW3 =  32 *  32, N3 = B * HW3;
  (void)HW2;

  float* act0  = alloc((size_t)64 * N0);   // 67 MB ping
  float* act1  = alloc((size_t)64 * N0);   // 67 MB pong
  float* ident = alloc((size_t)64 * N0);   // shortcut branch
  float* tmpC  = alloc((size_t)64 * N0);   // raw conv output (pre-BN)
  float* mid1  = alloc((size_t)16 * N0);   // bottleneck mid ping
  float* mid2  = alloc((size_t)16 * N0);   // bottleneck mid pong
  float* m1 = alloc(N1);
  float* m2 = alloc(N2);
  float* m3 = alloc(N3);
  float* stats = alloc(2 * 256);
  float* cnt   = alloc(4);

  // ---- launch helpers ----
  auto launch_conv = [&](const float* in, const float* w, float* out,
                         int Cin, int Cout, int Hs, int Ws, int K) {
    int Nsp = B * Hs * Ws;
    int Nt  = (Nsp + 15) / 16;
    int gx  = (Nt + 8 * NT - 1) / (8 * NT);   // 8 waves x NT tiles per block
    dim3 grid(gx, Cout / 16);
    size_t sh = (size_t)16 * Cin * K * K * sizeof(float);
    k_conv_wmma<<<grid, 256, sh, stream>>>(in, w, out, B, Cin, Cout, Hs, Ws, K,
                                           ilog2i(Hs * Ws), ilog2i(Ws));
  };
  auto launch_bn = [&](const float* cx, const float* m, const float* cntp,
                       const float* g, const float* bb, const float* resid,
                       float* out, int C, int Hs, int Ws, int relu) {
    int HWs = Hs * Ws, total = B * C * HWs;
    int lgC = ilog2i(C), lgHW = ilog2i(HWs);
    (void)hipMemsetAsync(stats, 0, 2 * C * sizeof(float), stream);
    int gx = (B * HWs + 255) / 256; if (gx > 256) gx = 256;
    k_bn_stats<<<dim3(gx, C), 256, 0, stream>>>(cx, m, stats, B, lgC, lgHW);
    k_bn_apply<<<(total + 255) / 256, 256, 0, stream>>>(cx, m, stats, cntp, g, bb,
                                                        resid, out, lgC, lgHW,
                                                        relu, total);
  };
  auto compute_cnt = [&](const float* m, int n, float* slot) {
    (void)hipMemsetAsync(slot, 0, sizeof(float), stream);
    int gx = (n + 255) / 256; if (gx > 1024) gx = 1024;
    k_sum<<<gx, 256, 0, stream>>>(m, slot, n);
  };
  auto run_bneck = [&](const BP& p, const float* xin, const float* m, const float* cntp,
                       float* out, int Hs, int Ws) {
    const float* id;
    if (p.has_s) {
      launch_conv(xin, p.wsd, tmpC, p.cin, p.cout, Hs, Ws, 1);
      launch_bn(tmpC, m, cntp, p.gs, p.bs, nullptr, ident, p.cout, Hs, Ws, 0);
      id = ident;
    } else id = xin;
    launch_conv(xin, p.w1, mid1, p.cin, p.mid, Hs, Ws, 1);
    launch_bn(mid1, m, cntp, p.g1, p.b1, nullptr, mid1, p.mid, Hs, Ws, 1);
    launch_conv(mid1, p.w2, mid2, p.mid, p.mid, Hs, Ws, p.k);
    launch_bn(mid2, m, cntp, p.g2, p.b2, nullptr, mid2, p.mid, Hs, Ws, 1);
    launch_conv(mid2, p.w3, tmpC, p.mid, p.cout, Hs, Ws, 1);
    launch_bn(tmpC, m, cntp, p.g3, p.b3, id, out, p.cout, Hs, Ws, 1);
  };
  auto run_pool = [&](const float* xin, const float* m, float* xout, float* nm,
                      int C, int Hs, int Ws) {
    int Ho = Hs / 2, Wo = Ws / 2;
    int lgH = ilog2i(Hs), lgW = ilog2i(Ws), lgHo = ilog2i(Ho), lgWo = ilog2i(Wo);
    int tmk = B * Ho * Wo;
    k_pool_mask<<<(tmk + 255) / 256, 256, 0, stream>>>(m, nm, Hs, Ws, lgH, lgW,
                                                       lgHo, lgWo, tmk);
    int tt = B * C * Ho * Wo;
    k_pool<<<(tt + 255) / 256, 256, 0, stream>>>(xin, m, nm, xout, Hs, Ws,
                                                 ilog2i(C), lgH, lgW, lgHo, lgWo, tt);
  };

  // ---- pipeline ----
  compute_cnt(mask0, N0, cnt + 0);
  {
    int t = B * 32 * HW0;
    k_mask_mul<<<(t + 255) / 256, 256, 0, stream>>>(x, mask0, act0, 5,
                                                    ilog2i(HW0), t);
  }
  run_bneck(b1a, act0, mask0, cnt + 0, act1, 256, 256);
  run_pool(act1, mask0, act0, m1, 64, 256, 256);
  compute_cnt(m1, N1, cnt + 1);
  run_bneck(b1b, act0, m1, cnt + 1, act1, 128, 128);
  run_bneck(b2a, act1, m1, cnt + 1, act0, 128, 128);
  run_pool(act0, m1, act1, m2, 128, 128, 128);
  compute_cnt(m2, N2, cnt + 2);
  run_bneck(b2b, act1, m2, cnt + 2, act0, 64, 64);
  run_bneck(b3a, act0, m2, cnt + 2, act1, 64, 64);
  run_pool(act1, m2, act0, m3, 256, 64, 64);
  compute_cnt(m3, N3, cnt + 3);
  run_bneck(b3b, act0, m3, cnt + 3, act1, 32, 32);

  launch_conv(act1, b4w, tmpC, 256, 256, 32, 32, 3);
  {
    int t = B * 256 * HW3;
    k_bias_mask<<<(t + 255) / 256, 256, 0, stream>>>(tmpC, b4b, m3, (float*)d_out,
                                                     8, ilog2i(HW3), t);
  }
}
